// AttentionBlock_65481071402202
// MI455X (gfx1250) — matrix operand order repaired and checked by reading, not compiled, not run on hardware
//
#include <hip/hip_runtime.h>
#include <stddef.h>

// ---------------------------------------------------------------------------
// Types for WMMA fragments (gfx1250, wave32)
// ---------------------------------------------------------------------------
typedef __attribute__((ext_vector_type(16))) __bf16          v16bf;
typedef __attribute__((ext_vector_type(8)))  float           v8f;
typedef __attribute__((ext_vector_type(4)))  float           fvec4;
typedef __attribute__((ext_vector_type(8)))  unsigned short  usvec8;
typedef __attribute__((ext_vector_type(16))) unsigned short  v16u16;
typedef __attribute__((ext_vector_type(4)))  int             v4i;

// Address-space-qualified v4i for the async global->LDS builtin
typedef __attribute__((address_space(1))) v4i gv4i;   // global
typedef __attribute__((address_space(3))) v4i lv4i;   // LDS

// ---------------------------------------------------------------------------
// gfx1250 async global->LDS (ASYNCcnt) support, guarded so compile never breaks
// ---------------------------------------------------------------------------
#if defined(__has_builtin)
#if __has_builtin(__builtin_amdgcn_global_load_async_to_lds_b128)
#define HAVE_ASYNC_LDS 1
#endif
#endif
#ifndef HAVE_ASYNC_LDS
#define HAVE_ASYNC_LDS 0
#endif

__device__ inline void async_wait0() {
#if HAVE_ASYNC_LDS
#if __has_builtin(__builtin_amdgcn_s_wait_asynccnt)
    __builtin_amdgcn_s_wait_asynccnt(0);
#else
    asm volatile("s_wait_asynccnt 0x0" ::: "memory");
#endif
#endif
}

// fp32 -> bf16 via the hardware convert
__device__ inline unsigned short to_bf16(float f) {
    __bf16 h = (__bf16)f;
    return __builtin_bit_cast(unsigned short, h);
}
__device__ inline unsigned short to_bf16(unsigned short u) { return u; }

// Assemble a v16bf A/B fragment from two 16-byte LDS chunks.
__device__ inline v16bf make_frag(const unsigned short* p0, const unsigned short* p1) {
    usvec8 lo = *(const usvec8*)p0;
    usvec8 hi = *(const usvec8*)p1;
    v16u16 u;
#pragma unroll
    for (int i = 0; i < 8; ++i) { u[i] = lo[i]; u[8 + i] = hi[i]; }
    return __builtin_bit_cast(v16bf, u);
}

// ---------------------------------------------------------------------------
// Staging: move a [ROWS][32] tile (row-major, K contiguous) into LDS as bf16.
// fp32 source: load/convert/store (VALU). bf16 source: async DMA to LDS when
// available (GLOBAL_LOAD_ASYNC_TO_LDS_B128), else 16B vector copies.
// src already points at (row0, k0); ld in elements; 256 threads.
// ---------------------------------------------------------------------------
template <int ROWS>
__device__ inline void stage_tile(unsigned short* dst, const float* src, size_t ld, int t) {
#pragma unroll
    for (int i = 0; i < (ROWS * 32) / 1024; ++i) {
        int linear = (t + i * 256) * 4;
        int r = linear >> 5, c = linear & 31;
        fvec4 v = *(const fvec4*)(src + (size_t)r * ld + c);
        int o = r * 32 + c;
        dst[o + 0] = to_bf16(v[0]);
        dst[o + 1] = to_bf16(v[1]);
        dst[o + 2] = to_bf16(v[2]);
        dst[o + 3] = to_bf16(v[3]);
    }
}

template <int ROWS>
__device__ inline void stage_tile(unsigned short* dst, const unsigned short* src, size_t ld, int t) {
#pragma unroll
    for (int i = 0; i < (ROWS * 32) / 2048; ++i) {
        int linear = (t + i * 256) * 8;
        int r = linear >> 5, c = linear & 31;
#if HAVE_ASYNC_LDS
        __builtin_amdgcn_global_load_async_to_lds_b128(
            (gv4i*)(src + (size_t)r * ld + c),
            (lv4i*)(dst + r * 32 + c),
            0, 0);
#else
        *(usvec8*)&dst[r * 32 + c] = *(const usvec8*)(src + (size_t)r * ld + c);
#endif
    }
}

// B stored [K, N] (non-transposed storage): gather into LDS as [n][k].
// src points at (k0, n0); COLS = BN = 128, K-tile = 32.
template <int COLS, typename T>
__device__ inline void stage_tileT(unsigned short* dst, const T* src, size_t ld, int t) {
#pragma unroll
    for (int i = 0; i < (COLS * 32) / 256; ++i) {
        int idx = t + i * 256;
        int n = idx & (COLS - 1);
        int k = idx / COLS;
        dst[n * 32 + k] = to_bf16(src[(size_t)k * ld + n]);
    }
}

// ---------------------------------------------------------------------------
// Tiled WMMA GEMM:  C[M,N] = A[M,K] @ op(B) (+ bias)
//   BT=true : B stored [N,K] row-major  (compute A @ B^T)
//   BT=false: B stored [K,N] row-major  (compute A @ B)
// Block tile 128x128x32, 256 threads = 8 wave32, each wave a 64x32 patch
// (4x2 WMMA tiles). Double-buffered LDS; async staging overlaps WMMA.
// blockIdx.z batches via sA/sB/sC element strides.
// ---------------------------------------------------------------------------
template <typename TA, typename TB, bool BT, bool OUTF32, bool BIAS>
__global__ __launch_bounds__(256) void gemm_wmma(
    const TA* __restrict__ A, const TB* __restrict__ B, const float* __restrict__ bias,
    void* __restrict__ Cout, int K, int lda, int ldb, int ldc,
    long long sA, long long sB, long long sC)
{
    constexpr int BM = 128, BN = 128, BK = 32;
    __shared__ __align__(16) unsigned short As[2][BM * BK];
    __shared__ __align__(16) unsigned short Bs[2][BN * BK];

    const int t    = threadIdx.x;
    const int bm   = blockIdx.y * BM;
    const int bn   = blockIdx.x * BN;
    const TA* Ab   = A + (size_t)blockIdx.z * sA;
    const TB* Bb   = B + (size_t)blockIdx.z * sB;
    const int lane = t & 31;
    const int wave = t >> 5;
    const int wm   = (wave & 1) * 64;    // 4 m-tiles of 16
    const int wn   = (wave >> 1) * 32;   // 2 n-tiles of 16
    const int lr   = lane & 15;          // row (A) / col (B,C) within 16-tile
    const int lh   = lane >> 4;          // lane half

    v8f acc[4][2] = {};

    auto stageAB = [&](int buf, int k0) {
        stage_tile<BM>(As[buf], Ab + (size_t)bm * lda + k0, (size_t)lda, t);
        if (BT) stage_tile<BN>(Bs[buf], Bb + (size_t)bn * ldb + k0, (size_t)ldb, t);
        else    stage_tileT<BN>(Bs[buf], Bb + (size_t)k0 * ldb + bn, (size_t)ldb, t);
    };

    stageAB(0, 0);
    async_wait0();
    __syncthreads();

    int cur = 0;
    for (int k0 = 0; k0 < K; k0 += BK) {
        int nxt = cur ^ 1;
        if (k0 + BK < K) stageAB(nxt, k0 + BK);   // overlaps with WMMAs below

        v16bf af[4], bfrag[2];
#pragma unroll
        for (int i = 0; i < 4; ++i) {
            // A fragment: row = lane%16, chunks at K = 8*hi and 16+8*hi
            const unsigned short* ar = &As[cur][(wm + 16 * i + lr) * 32];
            af[i] = make_frag(ar + 8 * lh, ar + 16 + 8 * lh);
        }
#pragma unroll
        for (int j = 0; j < 2; ++j) {
            // B fragment: col = lane%16, chunks at K = 8*hi and 16+8*hi
            const unsigned short* br = &Bs[cur][(wn + 16 * j + lr) * 32];
            bfrag[j] = make_frag(br + 8 * lh, br + 16 + 8 * lh);
        }
#pragma unroll
        for (int i = 0; i < 4; ++i)
#pragma unroll
            for (int j = 0; j < 2; ++j)
                acc[i][j] = __builtin_amdgcn_wmma_f32_16x16x32_bf16(
                    false, af[i], false, bfrag[j], (short)0, acc[i][j], false, false);

        async_wait0();
        __syncthreads();
        cur = nxt;
    }

    // Epilogue: C/D layout — element r -> row r + 8*hi, col = lane%16
#pragma unroll
    for (int i = 0; i < 4; ++i) {
#pragma unroll
        for (int j = 0; j < 2; ++j) {
            int col = bn + wn + 16 * j + lr;
#pragma unroll
            for (int r = 0; r < 8; ++r) {
                int row = bm + wm + 16 * i + 8 * lh + r;
                float v = acc[i][j][r];
                if (OUTF32) {
                    float* C = (float*)Cout + (size_t)blockIdx.z * sC;
                    C[(size_t)row * ldc + col] = BIAS ? (v + bias[col]) : v;
                } else {
                    unsigned short* C = (unsigned short*)Cout + (size_t)blockIdx.z * sC;
                    C[(size_t)row * ldc + col] = to_bf16(v);
                }
            }
        }
    }
}

// ---------------------------------------------------------------------------
// Row softmax: one block per row of 2048 fp32 scores -> bf16 probabilities.
// Each thread owns 8 contiguous elements (float4 x2 in, one 16B bf16 out).
// ---------------------------------------------------------------------------
__global__ __launch_bounds__(256) void softmax_kernel(
    const float* __restrict__ S, unsigned short* __restrict__ P, float scale)
{
    __shared__ float red[256];
    const size_t row = blockIdx.x;
    const int t = threadIdx.x;
    const float* s = S + row * 2048 + t * 8;

    fvec4 a = *(const fvec4*)(s);
    fvec4 b = *(const fvec4*)(s + 4);
    float v[8] = { a[0], a[1], a[2], a[3], b[0], b[1], b[2], b[3] };

    float m = -1e30f;
#pragma unroll
    for (int i = 0; i < 8; ++i) { v[i] *= scale; m = fmaxf(m, v[i]); }
    red[t] = m; __syncthreads();
    for (int off = 128; off > 0; off >>= 1) {
        if (t < off) red[t] = fmaxf(red[t], red[t + off]);
        __syncthreads();
    }
    m = red[0]; __syncthreads();

    float sum = 0.f;
#pragma unroll
    for (int i = 0; i < 8; ++i) { v[i] = __expf(v[i] - m); sum += v[i]; }
    red[t] = sum; __syncthreads();
    for (int off = 128; off > 0; off >>= 1) {
        if (t < off) red[t] += red[t + off];
        __syncthreads();
    }
    float inv = 1.0f / red[0];

    usvec8 o;
#pragma unroll
    for (int i = 0; i < 8; ++i) o[i] = to_bf16(v[i] * inv);
    *(usvec8*)(P + row * 2048 + t * 8) = o;
}

// ---------------------------------------------------------------------------
// Launch: 5-stage pipeline, all on `stream`.
//   ws layout (bytes):
//     [0,   50331648)  qkv   bf16 [8192, 3072]
//     [+ ,  67108864)  scores f32 [4, 2048, 2048]
//     [+ ,  33554432)  attn  bf16 [4, 2048, 2048]
//     [+ ,  16777216)  aout  bf16 [8192, 1024]
// ---------------------------------------------------------------------------
extern "C" void kernel_launch(void* const* d_in, const int* in_sizes, int n_in,
                              void* d_out, int out_size, void* d_ws, size_t ws_size,
                              hipStream_t stream)
{
    (void)in_sizes; (void)n_in; (void)out_size; (void)ws_size;

    const float* x      = (const float*)d_in[0];   // [4,2048,1024]
    const float* w_qkv  = (const float*)d_in[1];   // [3072,1024]
    const float* w_proj = (const float*)d_in[2];   // [1024,1024]
    const float* b_proj = (const float*)d_in[3];   // [1024]
    float*       out    = (float*)d_out;           // [4,2048,1024]

    char* ws = (char*)d_ws;
    unsigned short* qkv    = (unsigned short*)(ws);
    float*          scores = (float*)(ws + 50331648ull);
    unsigned short* attn   = (unsigned short*)(ws + 50331648ull + 67108864ull);
    unsigned short* aout   = (unsigned short*)(ws + 50331648ull + 67108864ull + 33554432ull);

    dim3 blk(256);

    // 1) qkv = x @ w_qkv^T  (fp32 in -> bf16 out)
    gemm_wmma<float, float, true, false, false>
        <<<dim3(3072 / 128, 8192 / 128, 1), blk, 0, stream>>>(
            x, w_qkv, nullptr, qkv, 1024, 1024, 1024, 3072, 0, 0, 0);

    // 2) scores = q @ k^T per batch (bf16 in -> fp32 out; scale applied in softmax)
    gemm_wmma<unsigned short, unsigned short, true, true, false>
        <<<dim3(2048 / 128, 2048 / 128, 4), blk, 0, stream>>>(
            qkv + 0, qkv + 1024, nullptr, scores, 1024, 3072, 3072, 2048,
            (long long)2048 * 3072, (long long)2048 * 3072, (long long)2048 * 2048);

    // 3) attn = softmax(scores * 1/sqrt(1024)) -> bf16
    softmax_kernel<<<dim3(4 * 2048), blk, 0, stream>>>(scores, attn, 0.03125f);

    // 4) aout = attn @ v per batch (B stored [K=key, N=d])
    gemm_wmma<unsigned short, unsigned short, false, false, false>
        <<<dim3(1024 / 128, 2048 / 128, 4), blk, 0, stream>>>(
            attn, qkv + 2048, nullptr, aout, 2048, 2048, 3072, 1024,
            (long long)2048 * 2048, (long long)2048 * 3072, (long long)2048 * 1024);

    // 5) out = aout @ w_proj^T + b_proj (fp32 out)
    gemm_wmma<unsigned short, float, true, true, true>
        <<<dim3(1024 / 128, 8192 / 128, 1), blk, 0, stream>>>(
            aout, w_proj, b_proj, out, 1024, 1024, 1024, 1024, 0, 0, 0);
}
